// GAT1_56478819943006
// MI455X (gfx1250) — matrix operand order repaired and checked by reading, not compiled, not run on hardware
//
#include <hip/hip_runtime.h>
#include <hip/hip_bf16.h>
#include <math.h>

#define GN 100000
#define GE 3200000
#define GF 128
#define GH 32
#define GB 256
#define NEG_SLOPE 0.2f

typedef __attribute__((ext_vector_type(16))) _Float16 v16h;
typedef __attribute__((ext_vector_type(8)))  float    v8f;

// ---------------------------------------------------------------------------
// Hardware atomics via CDNA5 inline asm (non-returning, device scope).
// global_atomic_add_f32 / global_atomic_max_num_f32: single L2-side RMW,
// tracked on STOREcnt; s_endpgm's implicit wait covers completion.
// ---------------------------------------------------------------------------
__device__ inline void atomicAddF32(float* addr, float v) {
    asm volatile("global_atomic_add_f32 %0, %1, off scope:SCOPE_DEV"
                 :: "v"(addr), "v"(v) : "memory");
}
__device__ inline void atomicMaxF32(float* addr, float v) {
    asm volatile("global_atomic_max_num_f32 %0, %1, off scope:SCOPE_DEV"
                 :: "v"(addr), "v"(v) : "memory");
}

// ---------------------------------------------------------------------------
// WMMA fragment loaders per CDNA5 ISA 7.12.2 layouts (wave32).
// A: 16x32 f16, M = lane&15; lanes 0-15 hold K = k0+{0..7, 16..23},
//    lanes 16-31 hold K = k0+{8..15, 24..31}.
// B: 32x16 f16, N = lane&15; lanes 0-15 hold K = k0+{0..7, 16..23},
//    lanes 16-31 hold K = k0+{8..15, 24..31}.
// D: v8f, element v -> row = v + 8*(lane>>4), col = lane&15.
// ---------------------------------------------------------------------------
__device__ inline v16h load_a_frag_f32(const float* A, int lda, int m0, int k0, int lane) {
    int row = m0 + (lane & 15);
    int kb  = k0 + ((lane >> 4) << 3);
    const float4* p4 = (const float4*)(A + (size_t)row * lda + kb);
    float4 q0 = p4[0], q1 = p4[1];   // K = kb+0..7
    float4 q2 = p4[4], q3 = p4[5];   // K = kb+16..23
    v16h a;
    a[0] = (_Float16)q0.x;  a[1] = (_Float16)q0.y;  a[2]  = (_Float16)q0.z;  a[3]  = (_Float16)q0.w;
    a[4] = (_Float16)q1.x;  a[5] = (_Float16)q1.y;  a[6]  = (_Float16)q1.z;  a[7]  = (_Float16)q1.w;
    a[8] = (_Float16)q2.x;  a[9] = (_Float16)q2.y;  a[10] = (_Float16)q2.z;  a[11] = (_Float16)q2.w;
    a[12] = (_Float16)q3.x; a[13] = (_Float16)q3.y; a[14] = (_Float16)q3.z;  a[15] = (_Float16)q3.w;
    return a;
}

__device__ inline v16h load_b_frag_f32(const float* Bw, int ldb, int k0, int n0, int lane) {
    int col = n0 + (lane & 15);
    int kb  = k0 + ((lane >> 4) << 3);
    const float* p = Bw + (size_t)kb * ldb + col;
    v16h b;
#pragma unroll
    for (int i = 0; i < 16; ++i) b[i] = (_Float16)p[(size_t)(i < 8 ? i : i + 8) * ldb];
    return b;
}

// ---------------------------------------------------------------------------
// Fused node transform: xl = x@Wl, xr = x@Wr (M=100000, K=128, N=32).
// One 16-row tile per wave; A fragment reused by 4 WMMAs per K step.
// 100000 = 6250 * 16 exactly -> no row tail. Guard is wave-uniform.
// ---------------------------------------------------------------------------
__global__ void node_transform_kernel(const float* __restrict__ x,
                                      const float* __restrict__ Wl,
                                      const float* __restrict__ Wr,
                                      float* __restrict__ xl,
                                      float* __restrict__ xr) {
    int wave = (int)((blockIdx.x * blockDim.x + threadIdx.x) >> 5);
    int lane = (int)(threadIdx.x & 31);
    if (wave >= GN / 16) return;  // uniform across wave (EXEC all-1 for WMMA)
    int m0 = wave << 4;

    v8f al0 = {}, al1 = {}, ar0 = {}, ar1 = {};
#pragma unroll
    for (int k0 = 0; k0 < GF; k0 += 32) {
        v16h a   = load_a_frag_f32(x, GF, m0, k0, lane);
        v16h bl0 = load_b_frag_f32(Wl, GH, k0, 0,  lane);
        v16h bl1 = load_b_frag_f32(Wl, GH, k0, 16, lane);
        v16h br0 = load_b_frag_f32(Wr, GH, k0, 0,  lane);
        v16h br1 = load_b_frag_f32(Wr, GH, k0, 16, lane);
        al0 = __builtin_amdgcn_wmma_f32_16x16x32_f16(false, a, false, bl0, (short)0, al0, false, false);
        al1 = __builtin_amdgcn_wmma_f32_16x16x32_f16(false, a, false, bl1, (short)0, al1, false, false);
        ar0 = __builtin_amdgcn_wmma_f32_16x16x32_f16(false, a, false, br0, (short)0, ar0, false, false);
        ar1 = __builtin_amdgcn_wmma_f32_16x16x32_f16(false, a, false, br1, (short)0, ar1, false, false);
    }

    int nlo = lane & 15;
    int mb  = m0 + ((lane >> 4) << 3);
#pragma unroll
    for (int v = 0; v < 8; ++v) {
        size_t rowoff = (size_t)(mb + v) * GH;
        xl[rowoff + nlo]      = al0[v];
        xl[rowoff + 16 + nlo] = al1[v];
        xr[rowoff + nlo]      = ar0[v];
        xr[rowoff + 16 + nlo] = ar1[v];
    }
}

// ---------------------------------------------------------------------------
// Generic C[M,N] = act(A[M,K] @ B[K,N] + bias). One 16x16 tile per wave.
// Used for the MLP head layers (M=256; N=1024/512; K=32/1024).
// ---------------------------------------------------------------------------
__global__ void wmma_gemm_kernel(const float* __restrict__ A,
                                 const float* __restrict__ Bw,
                                 const float* __restrict__ bias,
                                 float* __restrict__ C,
                                 int M, int N, int K, int relu) {
    int wave = (int)((blockIdx.x * blockDim.x + threadIdx.x) >> 5);
    int lane = (int)(threadIdx.x & 31);
    int tiles_n = N >> 4;
    int tiles_m = M >> 4;
    if (wave >= tiles_m * tiles_n) return;  // uniform across wave
    int m0 = (wave / tiles_n) << 4;
    int n0 = (wave % tiles_n) << 4;

    v8f acc = {};
    for (int k0 = 0; k0 < K; k0 += 32) {
        v16h a = load_a_frag_f32(A, K, m0, k0, lane);
        v16h b = load_b_frag_f32(Bw, N, k0, n0, lane);
        acc = __builtin_amdgcn_wmma_f32_16x16x32_f16(false, a, false, b,
                                                     (short)0, acc, false, false);
    }

    int n  = n0 + (lane & 15);
    int mb = m0 + ((lane >> 4) << 3);
    float bv = bias ? bias[n] : 0.0f;
#pragma unroll
    for (int v = 0; v < 8; ++v) {
        float val = acc[v] + bv;
        if (relu) val = fmaxf(val, 0.0f);
        C[(size_t)(mb + v) * N + n] = val;
    }
}

// ---------------------------------------------------------------------------
// Init: m = -inf, z = 0, node_out = bias (broadcast), g = -inf
// ---------------------------------------------------------------------------
__global__ void init_kernel(float* __restrict__ m, float* __restrict__ z,
                            float* __restrict__ node_out, float* __restrict__ g,
                            const float* __restrict__ bias) {
    size_t i = (size_t)blockIdx.x * blockDim.x + threadIdx.x;
    if (i < (size_t)GN) { m[i] = -INFINITY; z[i] = 0.0f; }
    if (i < (size_t)GN * GH) node_out[i] = bias[i & (GH - 1)];
    if (i < (size_t)GB * GH) g[i] = -INFINITY;
}

// ---------------------------------------------------------------------------
// Edge stage 1: e = leaky_relu(xl[src] + xr[dst]) . att ; m[dst] = max(m, e)
// xl/xr are 12.8 MB each -> L2-resident gathers.
// ---------------------------------------------------------------------------
__global__ void edge_score_kernel(const float* __restrict__ xl,
                                  const float* __restrict__ xr,
                                  const int* __restrict__ ei,
                                  const float* __restrict__ att,
                                  float* __restrict__ e,
                                  float* __restrict__ m) {
    int i = (int)(blockIdx.x * blockDim.x + threadIdx.x);
    const int EN = GE + GN;
    if (i >= EN) return;
    int src = (i < GE) ? ei[i] : (i - GE);
    int dst = (i < GE) ? ei[GE + i] : (i - GE);
    const float4* pl = (const float4*)(xl + (size_t)src * GH);
    const float4* pr = (const float4*)(xr + (size_t)dst * GH);
    const float4* pa = (const float4*)att;
    float acc = 0.0f;
#pragma unroll
    for (int j = 0; j < GH / 4; ++j) {
        float4 a = pl[j], b = pr[j], w = pa[j];
        float s0 = a.x + b.x; s0 = (s0 > 0.0f) ? s0 : NEG_SLOPE * s0;
        float s1 = a.y + b.y; s1 = (s1 > 0.0f) ? s1 : NEG_SLOPE * s1;
        float s2 = a.z + b.z; s2 = (s2 > 0.0f) ? s2 : NEG_SLOPE * s2;
        float s3 = a.w + b.w; s3 = (s3 > 0.0f) ? s3 : NEG_SLOPE * s3;
        acc += s0 * w.x + s1 * w.y + s2 * w.z + s3 * w.w;
    }
    e[i] = acc;
    atomicMaxF32(&m[dst], acc);
}

// ---------------------------------------------------------------------------
// Edge stage 2: p = exp(e - m[dst]) (in place over e); z[dst] += p
// ---------------------------------------------------------------------------
__global__ void edge_exp_kernel(const int* __restrict__ ei,
                                float* __restrict__ e,
                                const float* __restrict__ m,
                                float* __restrict__ z) {
    int i = (int)(blockIdx.x * blockDim.x + threadIdx.x);
    const int EN = GE + GN;
    if (i >= EN) return;
    int dst = (i < GE) ? ei[GE + i] : (i - GE);
    float p = __expf(e[i] - m[dst]);
    e[i] = p;
    atomicAddF32(&z[dst], p);
}

// ---------------------------------------------------------------------------
// Edge stage 3: node_out[dst] += (p/z[dst]) * xl[src]
// ---------------------------------------------------------------------------
__global__ void edge_scatter_kernel(const int* __restrict__ ei,
                                    const float* __restrict__ e,
                                    const float* __restrict__ z,
                                    const float* __restrict__ xl,
                                    float* __restrict__ node_out) {
    int i = (int)(blockIdx.x * blockDim.x + threadIdx.x);
    const int EN = GE + GN;
    if (i >= EN) return;
    int src = (i < GE) ? ei[i] : (i - GE);
    int dst = (i < GE) ? ei[GE + i] : (i - GE);
    float alpha = e[i] / z[dst];
    const float4* pl = (const float4*)(xl + (size_t)src * GH);
    float* po = node_out + (size_t)dst * GH;
#pragma unroll
    for (int j = 0; j < GH / 4; ++j) {
        float4 a = pl[j];
        atomicAddF32(&po[4 * j + 0], alpha * a.x);
        atomicAddF32(&po[4 * j + 1], alpha * a.y);
        atomicAddF32(&po[4 * j + 2], alpha * a.z);
        atomicAddF32(&po[4 * j + 3], alpha * a.w);
    }
}

// ---------------------------------------------------------------------------
// Global max pool: g[batch[n]][h] = max(node_out[n][h])
// ---------------------------------------------------------------------------
__global__ void pool_kernel(const float* __restrict__ node_out,
                            const int* __restrict__ batch,
                            float* __restrict__ g) {
    size_t i = (size_t)blockIdx.x * blockDim.x + threadIdx.x;
    if (i >= (size_t)GN * GH) return;
    int node = (int)(i >> 5);
    int h    = (int)(i & (GH - 1));
    atomicMaxF32(&g[(size_t)batch[node] * GH + h], node_out[i]);
}

// ---------------------------------------------------------------------------
// MLP layer 3 (N=4, too narrow for WMMA): out = h2 @ W3 + b3
// ---------------------------------------------------------------------------
__global__ void mlp3_kernel(const float* __restrict__ h2,
                            const float* __restrict__ W3,
                            const float* __restrict__ b3,
                            float* __restrict__ out) {
    int i = (int)(blockIdx.x * blockDim.x + threadIdx.x);
    if (i >= GB * 4) return;
    int r = i >> 2, c = i & 3;
    float acc = b3[c];
    const float* hr = h2 + (size_t)r * 512;
    for (int k = 0; k < 512; ++k) acc += hr[k] * W3[(size_t)k * 4 + c];
    out[i] = acc;
}

// ---------------------------------------------------------------------------
extern "C" void kernel_launch(void* const* d_in, const int* in_sizes, int n_in,
                              void* d_out, int out_size, void* d_ws, size_t ws_size,
                              hipStream_t stream) {
    const float* x    = (const float*)d_in[0];
    const int*   ei   = (const int*)  d_in[1];
    const int*   bat  = (const int*)  d_in[2];
    const float* Wl   = (const float*)d_in[3];
    const float* Wr   = (const float*)d_in[4];
    const float* att  = (const float*)d_in[5];
    const float* bias = (const float*)d_in[6];
    const float* W1   = (const float*)d_in[7];
    const float* b1   = (const float*)d_in[8];
    const float* W2   = (const float*)d_in[9];
    const float* b2   = (const float*)d_in[10];
    const float* W3   = (const float*)d_in[11];
    const float* b3   = (const float*)d_in[12];
    float* out = (float*)d_out;

    // workspace carve-up (floats)
    float* ws = (float*)d_ws;
    size_t off = 0;
    float* xl    = ws + off; off += (size_t)GN * GH;
    float* xr    = ws + off; off += (size_t)GN * GH;
    float* ebuf  = ws + off; off += (size_t)(GE + GN);   // scores, then p in-place
    float* mmax  = ws + off; off += (size_t)GN;
    float* zsum  = ws + off; off += (size_t)GN;
    float* nodeO = ws + off; off += (size_t)GN * GH;
    float* g     = ws + off; off += (size_t)GB * GH;
    float* h1    = ws + off; off += (size_t)GB * 1024;
    float* h2    = ws + off; off += (size_t)GB * 512;
    (void)ws_size;

    const int T = 256;
    const int EN = GE + GN;

    // 0) init m/z/node_out/g
    {
        size_t total = (size_t)GN * GH;
        init_kernel<<<(int)((total + T - 1) / T), T, 0, stream>>>(mmax, zsum, nodeO, g, bias);
    }

    // 1) fused xl = x@Wl, xr = x@Wr (WMMA, x streamed once)
    {
        int waves  = GN / 16;                       // 6250 tiles
        int blocks = (waves * 32 + T - 1) / T;
        node_transform_kernel<<<blocks, T, 0, stream>>>(x, Wl, Wr, xl, xr);
    }

    // 2) edge scores + segment max (hw f32 max atomics)
    edge_score_kernel<<<(EN + T - 1) / T, T, 0, stream>>>(xl, xr, ei, att, ebuf, mmax);

    // 3) exp + segment sum (hw f32 add atomics)
    edge_exp_kernel<<<(EN + T - 1) / T, T, 0, stream>>>(ei, ebuf, mmax, zsum);

    // 4) weighted scatter-sum into node_out (pre-initialized with bias)
    edge_scatter_kernel<<<(EN + T - 1) / T, T, 0, stream>>>(ei, ebuf, zsum, xl, nodeO);

    // 5) global max pool per graph
    {
        size_t total = (size_t)GN * GH;
        pool_kernel<<<(int)((total + T - 1) / T), T, 0, stream>>>(nodeO, bat, g);
    }

    // 6) MLP head: h1 = relu(g@W1+b1), h2 = relu(h1@W2+b2)  (WMMA)
    {
        int waves  = (GB / 16) * (1024 / 16);
        int blocks = (waves * 32 + T - 1) / T;
        wmma_gemm_kernel<<<blocks, T, 0, stream>>>(g, W1, b1, h1, GB, 1024, GH, 1);
    }
    {
        int waves  = (GB / 16) * (512 / 16);
        int blocks = (waves * 32 + T - 1) / T;
        wmma_gemm_kernel<<<blocks, T, 0, stream>>>(h1, W2, b2, h2, GB, 512, 1024, 1);
    }

    // 7) final layer -> d_out [256,4]
    mlp3_kernel<<<(GB * 4 + T - 1) / T, T, 0, stream>>>(h2, W3, b3, out);
}
